// MultiHeadAttention_33844342293035
// MI455X (gfx1250) — matrix operand order repaired and checked by reading, not compiled, not run on hardware
//
#include <hip/hip_runtime.h>
#include <hip/hip_bf16.h>
#include <stdint.h>

#define EMB 512
#define HEADS 8
#define HDIM 64
#define SEQ 2048
#define BATCH 4
#define ROWS (BATCH * SEQ)  // 8192

typedef __attribute__((ext_vector_type(16))) __bf16 v16bf;
typedef __attribute__((ext_vector_type(8)))  float  v8f;
typedef int v4i __attribute__((vector_size(16)));

#if defined(__has_builtin)
#if __has_builtin(__builtin_amdgcn_global_load_async_to_lds_b128)
#define HAVE_ASYNC 1
#endif
#endif
#ifndef HAVE_ASYNC
#define HAVE_ASYNC 0
#endif

union Frag {
    v16bf v;
    uint4 q[2];
    unsigned short s[16];
};

static __device__ inline unsigned short f2bf(float f) {
    union { float f; unsigned int u; } c; c.f = f;
    unsigned int u = c.u;
    unsigned int r = u + 0x7FFFu + ((u >> 16) & 1u);  // round-to-nearest-even
    return (unsigned short)(r >> 16);
}

// Async 16-byte global->LDS copy (falls back to load+ds_store when builtin absent).
static __device__ inline void cp_b128(const unsigned short* g, unsigned short* l) {
#if HAVE_ASYNC
    typedef __attribute__((address_space(1))) v4i gv4i;
    typedef __attribute__((address_space(3))) v4i lv4i;
    __builtin_amdgcn_global_load_async_to_lds_b128(
        (gv4i*)(uintptr_t)g, (lv4i*)(uintptr_t)l, 0, 0);
#else
    *(uint4*)l = *(const uint4*)g;
#endif
}

static __device__ inline void cp_wait_barrier() {
#if HAVE_ASYNC
    asm volatile("s_wait_asynccnt 0x0" ::: "memory");
#endif
    __syncthreads();
}

// A-fragment (16x32 bf16, row-major source): lane lo holds row, K = hi*16 + {0..15}
static __device__ inline v16bf loadA(const unsigned short* __restrict__ base, int ld,
                                     int row0, int k0, int lo, int hi) {
    Frag f;
    const unsigned short* p = base + (size_t)(row0 + lo) * ld + k0 + hi * 16;
    f.q[0] = *(const uint4*)(p);
    f.q[1] = *(const uint4*)(p + 8);
    return f.v;
}

// B-fragment from 16 contiguous bf16 (lane = column, K split 16/16 across lane halves).
static __device__ inline v16bf loadB16(const unsigned short* p) {
    Frag f;
    f.q[0] = *(const uint4*)(p);
    f.q[1] = *(const uint4*)(p + 8);
    return f.v;
}

__global__ void mha_cvt_bf16(const float* __restrict__ in, unsigned short* __restrict__ out, int n) {
    int i = blockIdx.x * blockDim.x + threadIdx.x;
    int stride = gridDim.x * blockDim.x;
    for (; i < n; i += stride) out[i] = f2bf(in[i]);
}

// ---------------------------------------------------------------------------
// Q/K/V projection: 128-thread block computes 64 rows x 64 cols.
// Weight tile (64 cols x 32 k) staged in LDS (double-buffered, async), shared by 4 waves.
// grid = (ROWS/64, 24): blockIdx.y/8 selects matrix, %8 selects 64-col slab.
// ---------------------------------------------------------------------------
__global__ __launch_bounds__(128) void mha_qkv_gemm(
    const unsigned short* __restrict__ xb,
    const unsigned short* __restrict__ wqb,
    const unsigned short* __restrict__ wkb,
    const unsigned short* __restrict__ wvb,
    const float* __restrict__ bq, const float* __restrict__ bk, const float* __restrict__ bv,
    unsigned short* __restrict__ Qb,   // [b,h,n,d]
    unsigned short* __restrict__ Kb,   // [b,h,n,d]
    unsigned short* __restrict__ Vtb)  // [b,h,d,n]
{
    __shared__ __align__(16) unsigned short Bt[2][64 * 32];
    int tid = threadIdx.x;
    int lane = tid & 31, lo = lane & 15, hi = lane >> 4, w = tid >> 5;
    int colgroup = blockIdx.y;
    int mat = colgroup >> 3;  // 0=Q 1=K 2=V
    int cg = colgroup & 7;
    const unsigned short* W = (mat == 0) ? wqb : ((mat == 1) ? wkb : wvb);
    const float* bias = (mat == 0) ? bq : ((mat == 1) ? bk : bv);
    int colbase = cg * 64;
    int row0 = blockIdx.x * 64 + w * 16;

    // stage weight tile [64 x 32] at k0 into Bt[buf]
    auto stage = [&](int k0, int buf) {
#pragma unroll
        for (int c2 = 0; c2 < 2; c2++) {
            int chunk = tid * 2 + c2;         // 0..255
            int wrow = chunk >> 2;            // 0..63
            int off = (chunk & 3) * 8;        // 0,8,16,24
            cp_b128(W + (size_t)(colbase + wrow) * EMB + k0 + off,
                    &Bt[buf][wrow * 32 + off]);
        }
    };

    stage(0, 0);
    v8f acc[4] = {};
    for (int k0 = 0; k0 < EMB; k0 += 32) {
        int buf = (k0 >> 5) & 1;
        cp_wait_barrier();
        if (k0 + 32 < EMB) stage(k0 + 32, buf ^ 1);
        v16bf a = loadA(xb, EMB, row0, k0, lo, hi);
#pragma unroll
        for (int c = 0; c < 4; c++) {
            v16bf b = loadB16(&Bt[buf][(c * 16 + lo) * 32 + hi * 16]);
            acc[c] = __builtin_amdgcn_wmma_f32_16x16x32_bf16(false, a, false, b,
                                                             (short)0, acc[c], false, false);
        }
    }
#pragma unroll
    for (int c = 0; c < 4; c++) {
        int col = colbase + c * 16 + lo;
        float bval = bias[col];
        int h = col >> 6, d = col & 63;
#pragma unroll
        for (int i = 0; i < 8; i++) {
            int m = row0 + hi * 8 + i;
            int bi = m >> 11, nn = m & 2047;
            unsigned short r = f2bf(acc[c][i] + bval);
            if (mat == 2) {
                Vtb[((size_t)(bi * HEADS + h) * HDIM + d) * SEQ + nn] = r;
            } else {
                unsigned short* dst = (mat == 0) ? Qb : Kb;
                dst[((size_t)(bi * HEADS + h) * SEQ + nn) * HDIM + d] = r;
            }
        }
    }
}

// ---------------------------------------------------------------------------
// Flash attention: 128-thread block = 4 waves = 64 query rows of one (b,h).
// K tile (32xHDIM) and V^T tile (HDIMx32) staged in LDS (double-buffered, async),
// shared by all 4 waves. Online softmax per wave; P restaged via per-wave LDS.
// grid = (BATCH*HEADS, SEQ/64)
// ---------------------------------------------------------------------------
__global__ __launch_bounds__(128) void mha_flash(
    const unsigned short* __restrict__ Qb,
    const unsigned short* __restrict__ Kb,
    const unsigned short* __restrict__ Vtb,
    unsigned short* __restrict__ ctx)  // [b,n,EMB] bf16
{
    __shared__ __align__(16) unsigned short Kt[2][32 * 64];
    __shared__ __align__(16) unsigned short Vt[2][64 * 32];
    __shared__ __align__(16) unsigned short Pl[4][16 * 32];
    int tid = threadIdx.x;
    int lane = tid & 31, lo = lane & 15, hi = lane >> 4, w = tid >> 5;
    int bh = blockIdx.x;                    // 0..31
    int qtile = blockIdx.y * 4 + w;         // 0..127
    const unsigned short* Qh = Qb + (size_t)bh * SEQ * HDIM;
    const unsigned short* Kh = Kb + (size_t)bh * SEQ * HDIM;
    const unsigned short* Vh = Vtb + (size_t)bh * HDIM * SEQ;

    v16bf qf0 = loadA(Qh, HDIM, qtile * 16, 0, lo, hi);
    v16bf qf1 = loadA(Qh, HDIM, qtile * 16, 32, lo, hi);

    // stage K rows [j..j+31] (32x64) and V^T cols (64x32) into buf
    auto stage = [&](int j, int buf) {
#pragma unroll
        for (int c2 = 0; c2 < 2; c2++) {
            int chunk = tid * 2 + c2;     // 0..255
            int krow = chunk >> 3;        // 0..31
            int koff = (chunk & 7) * 8;   // 0..56
            cp_b128(Kh + (size_t)(j + krow) * HDIM + koff, &Kt[buf][krow * 64 + koff]);
            int drow = chunk >> 2;        // 0..63
            int voff = (chunk & 3) * 8;   // 0..24
            cp_b128(Vh + (size_t)drow * SEQ + j + voff, &Vt[buf][drow * 32 + voff]);
        }
    };

    v8f O[4] = {};
    float M[8], L[8];
#pragma unroll
    for (int i = 0; i < 8; i++) { M[i] = -1e30f; L[i] = 0.f; }
    const float scale = 0.044194173824159216f;  // 1/sqrt(EMB)

    stage(0, 0);
    for (int j = 0; j < SEQ; j += 32) {
        int buf = (j >> 5) & 1;
        cp_wait_barrier();
        if (j + 32 < SEQ) stage(j + 32, buf ^ 1);

        v8f s0 = {}, s1 = {};
        {
            v16bf b00 = loadB16(&Kt[buf][lo * 64 + hi * 16]);
            v16bf b01 = loadB16(&Kt[buf][lo * 64 + 32 + hi * 16]);
            s0 = __builtin_amdgcn_wmma_f32_16x16x32_bf16(false, qf0, false, b00, (short)0, s0, false, false);
            s0 = __builtin_amdgcn_wmma_f32_16x16x32_bf16(false, qf1, false, b01, (short)0, s0, false, false);
            v16bf b10 = loadB16(&Kt[buf][(16 + lo) * 64 + hi * 16]);
            v16bf b11 = loadB16(&Kt[buf][(16 + lo) * 64 + 32 + hi * 16]);
            s1 = __builtin_amdgcn_wmma_f32_16x16x32_bf16(false, qf0, false, b10, (short)0, s1, false, false);
            s1 = __builtin_amdgcn_wmma_f32_16x16x32_bf16(false, qf1, false, b11, (short)0, s1, false, false);
        }
#pragma unroll
        for (int i = 0; i < 8; i++) {
            float a = s0[i] * scale, b = s1[i] * scale;
            float mx = fmaxf(a, b);
#pragma unroll
            for (int msk = 8; msk >= 1; msk >>= 1) mx = fmaxf(mx, __shfl_xor(mx, msk, 32));
            float Mn = fmaxf(M[i], mx);
            float alpha = __expf(M[i] - Mn);
            float p0 = __expf(a - Mn), p1 = __expf(b - Mn);
            float sm = p0 + p1;
#pragma unroll
            for (int msk = 8; msk >= 1; msk >>= 1) sm += __shfl_xor(sm, msk, 32);
            L[i] = L[i] * alpha + sm;
            M[i] = Mn;
#pragma unroll
            for (int t = 0; t < 4; t++) O[t][i] *= alpha;
            Pl[w][(hi * 8 + i) * 32 + lo] = f2bf(p0);
            Pl[w][(hi * 8 + i) * 32 + 16 + lo] = f2bf(p1);
        }
        asm volatile("s_wait_dscnt 0x0" ::: "memory");
        Frag pf;
        const unsigned short* pp = &Pl[w][lo * 32 + hi * 16];
        pf.q[0] = *(const uint4*)(pp);
        pf.q[1] = *(const uint4*)(pp + 8);
#pragma unroll
        for (int t = 0; t < 4; t++) {
            v16bf vb = loadB16(&Vt[buf][(t * 16 + lo) * 32 + hi * 16]);
            O[t] = __builtin_amdgcn_wmma_f32_16x16x32_bf16(false, pf.v, false, vb, (short)0, O[t], false, false);
        }
    }

    int bi = bh >> 3, h = bh & 7;
#pragma unroll
    for (int t = 0; t < 4; t++) {
#pragma unroll
        for (int i = 0; i < 8; i++) {
            float v = O[t][i] / L[i];
            int qrow = qtile * 16 + hi * 8 + i;
            int col = h * HDIM + t * 16 + lo;
            ctx[(size_t)(bi * SEQ + qrow) * EMB + col] = f2bf(v);
        }
    }
}

// ---------------------------------------------------------------------------
// Output projection: out = ctx @ Wo^T + bo (fp32 out). Same LDS-staged structure.
// grid = (ROWS/64, 8)
// ---------------------------------------------------------------------------
__global__ __launch_bounds__(128) void mha_out_proj(
    const unsigned short* __restrict__ ctx,
    const unsigned short* __restrict__ wob,
    const float* __restrict__ bo,
    float* __restrict__ out)
{
    __shared__ __align__(16) unsigned short Bt[2][64 * 32];
    int tid = threadIdx.x;
    int lane = tid & 31, lo = lane & 15, hi = lane >> 4, w = tid >> 5;
    int cg = blockIdx.y;
    int colbase = cg * 64;
    int row0 = blockIdx.x * 64 + w * 16;

    auto stage = [&](int k0, int buf) {
#pragma unroll
        for (int c2 = 0; c2 < 2; c2++) {
            int chunk = tid * 2 + c2;
            int wrow = chunk >> 2;
            int off = (chunk & 3) * 8;
            cp_b128(wob + (size_t)(colbase + wrow) * EMB + k0 + off,
                    &Bt[buf][wrow * 32 + off]);
        }
    };

    stage(0, 0);
    v8f acc[4] = {};
    for (int k0 = 0; k0 < EMB; k0 += 32) {
        int buf = (k0 >> 5) & 1;
        cp_wait_barrier();
        if (k0 + 32 < EMB) stage(k0 + 32, buf ^ 1);
        v16bf a = loadA(ctx, EMB, row0, k0, lo, hi);
#pragma unroll
        for (int c = 0; c < 4; c++) {
            v16bf b = loadB16(&Bt[buf][(c * 16 + lo) * 32 + hi * 16]);
            acc[c] = __builtin_amdgcn_wmma_f32_16x16x32_bf16(false, a, false, b,
                                                             (short)0, acc[c], false, false);
        }
    }
#pragma unroll
    for (int c = 0; c < 4; c++) {
        int col = colbase + c * 16 + lo;
        float bb = bo[col];
#pragma unroll
        for (int i = 0; i < 8; i++) {
            int m = row0 + hi * 8 + i;
            out[(size_t)m * EMB + col] = acc[c][i] + bb;
        }
    }
}

extern "C" void kernel_launch(void* const* d_in, const int* in_sizes, int n_in,
                              void* d_out, int out_size, void* d_ws, size_t ws_size,
                              hipStream_t stream) {
    const float* x  = (const float*)d_in[0];
    const float* Wq = (const float*)d_in[1];
    const float* bq = (const float*)d_in[2];
    const float* Wk = (const float*)d_in[3];
    const float* bk = (const float*)d_in[4];
    const float* Wv = (const float*)d_in[5];
    const float* bv = (const float*)d_in[6];
    const float* Wo = (const float*)d_in[7];
    const float* bo = (const float*)d_in[8];
    float* out = (float*)d_out;

    char* p = (char*)d_ws;
    unsigned short* xb  = (unsigned short*)p; p += (size_t)ROWS * EMB * 2;
    unsigned short* wqb = (unsigned short*)p; p += (size_t)EMB * EMB * 2;
    unsigned short* wkb = (unsigned short*)p; p += (size_t)EMB * EMB * 2;
    unsigned short* wvb = (unsigned short*)p; p += (size_t)EMB * EMB * 2;
    unsigned short* wob = (unsigned short*)p; p += (size_t)EMB * EMB * 2;
    unsigned short* Qb  = (unsigned short*)p; p += (size_t)ROWS * EMB * 2;
    unsigned short* Kb  = (unsigned short*)p; p += (size_t)ROWS * EMB * 2;
    unsigned short* Vtb = (unsigned short*)p; p += (size_t)ROWS * EMB * 2;
    unsigned short* ctx = (unsigned short*)p; p += (size_t)ROWS * EMB * 2;

    mha_cvt_bf16<<<4096, 256, 0, stream>>>(x,  xb,  ROWS * EMB);
    mha_cvt_bf16<<<512,  256, 0, stream>>>(Wq, wqb, EMB * EMB);
    mha_cvt_bf16<<<512,  256, 0, stream>>>(Wk, wkb, EMB * EMB);
    mha_cvt_bf16<<<512,  256, 0, stream>>>(Wv, wvb, EMB * EMB);
    mha_cvt_bf16<<<512,  256, 0, stream>>>(Wo, wob, EMB * EMB);

    dim3 g1(ROWS / 64, 24);
    mha_qkv_gemm<<<g1, 128, 0, stream>>>(xb, wqb, wkb, wvb, bq, bk, bv, Qb, Kb, Vtb);

    dim3 g2(BATCH * HEADS, SEQ / 64);
    mha_flash<<<g2, 128, 0, stream>>>(Qb, Kb, Vtb, ctx);

    dim3 g3(ROWS / 64, 8);
    mha_out_proj<<<g3, 128, 0, stream>>>(ctx, wob, bo, out);
}
